// RNN_GRU_39539468927124
// MI455X (gfx1250) — compile-verified, matrix operand order checked by reading, not run on hardware
//
#include <hip/hip_runtime.h>
#include <math.h>

// ---- problem constants (from reference) ----
constexpr int HID   = 128;
constexpr int DIN   = 72;
constexpr int KPAD  = 96;     // DIN padded to 3 K-steps of 32
constexpr int TIN   = 50;
constexpr int BATCH = 2048;
constexpr int ROWS  = 16;     // batch rows per workgroup
constexpr int NSTEPS = 99;    // 49 encoder + 50 decoder LSTM steps

typedef __attribute__((ext_vector_type(16))) _Float16 v16h;
typedef __attribute__((ext_vector_type(8)))  _Float16 v8h;
typedef __attribute__((ext_vector_type(8)))  float    v8f;

// Build a WMMA B-fragment (K=32 x N=16, f16) from a row-major f32 weight
// matrix W[n][k] (B[k][n] = W[n][k]).  ISA 7.12.2 16-bit layout:
// lane L holds column n = base + L%16; half j holds
// k = kbase + (L/16)*8 + (j<8 ? j : 8+j).
__device__ __forceinline__ v16h load_b_frag(const float* __restrict__ W,
                                            int n, int stride, int klim, int nlim,
                                            int kbase, int lane) {
  v16h b;
  const int l8 = (lane >> 4) << 3;
#pragma unroll
  for (int j = 0; j < 16; ++j) {
    const int k = kbase + l8 + ((j < 8) ? j : (8 + j));
    float v = 0.0f;
    if (k < klim && n < nlim) v = W[n * stride + k];
    b[j] = (_Float16)v;
  }
  return b;
}

// A-fragment (M=16 x K=32, f16) from an LDS tile, same K pattern:
// two 16B contiguous chunks per lane -> 2x ds_load_b128.
__device__ __forceinline__ v16h load_a_frag(const _Float16* base, int rowHalves,
                                            int kbase, int lane) {
  const int m  = lane & 15;
  const int k0 = kbase + ((lane >> 4) << 3);
  const _Float16* p = base + m * rowHalves + k0;
  v8h lo = *(const v8h*)(p);
  v8h hi = *(const v8h*)(p + 16);
  v16h a;
#pragma unroll
  for (int j = 0; j < 8; ++j) { a[j] = lo[j]; a[j + 8] = hi[j]; }
  return a;
}

__device__ __forceinline__ float sigmoid_fast(float x) {
  return 1.0f / (1.0f + __expf(-x));          // stable both directions
}
__device__ __forceinline__ float tanh_fast(float x) {
  const float ax = fabsf(x);
  const float t  = __expf(-2.0f * ax);        // in (0,1], no overflow
  const float r  = (1.0f - t) / (1.0f + t);
  return copysignf(r, x);
}

// One persistent workgroup per 16 batch rows; wave w owns gate w (i,f,g,o).
__global__ __launch_bounds__(128, 1) void lstm_seq2seq_kernel(
    const float* __restrict__ x,    // [50][2048][72]
    const float* __restrict__ Wih,  // [512][72]
    const float* __restrict__ Whh,  // [512][128]
    const float* __restrict__ bih,  // [512]
    const float* __restrict__ bhh,  // [512]
    const float* __restrict__ fcW,  // [72][128]
    const float* __restrict__ fcb,  // [72]
    float* __restrict__ out)        // [100][2048][72]
{
  __shared__ __align__(16) _Float16 sInp[ROWS][KPAD];   // current input tile (f16, zero-padded)
  __shared__ __align__(16) _Float16 sH[ROWS][HID];      // hidden state (f16 for WMMA)
  __shared__ __align__(16) float    sC[ROWS][HID];      // cell state (f32)
  __shared__ __align__(16) float    sG[4][ROWS][HID];   // post-activation gates (f32)

  const int tid  = threadIdx.x;
  const int lane = tid & 31;
  const int wave = tid >> 5;              // gate index
  const int row0 = blockIdx.x * ROWS;
  const int col  = lane & 15;
  const int mhi  = (lane >> 4) << 3;      // +8 row offset for lanes 16..31 in C/D layout

  // ---- init LDS state ----
  for (int i = tid; i < ROWS * KPAD; i += 128) ((_Float16*)sInp)[i] = (_Float16)0.0f;
  for (int i = tid; i < ROWS * HID; i += 128) {
    ((_Float16*)sH)[i] = (_Float16)0.0f;
    ((float*)sC)[i]    = 0.0f;
  }
  __syncthreads();
  // first input = x[0]
  for (int i = tid; i < ROWS * DIN; i += 128) {
    const int m = i / DIN, d = i % DIN;
    sInp[m][d] = (_Float16)x[(0 * BATCH + row0 + m) * DIN + d];
  }

  // ---- weight-stationary fragments (live in VGPRs for all 99 steps) ----
  v16h bWih[3][8];
#pragma unroll
  for (int ks = 0; ks < 3; ++ks)
#pragma unroll
    for (int nt = 0; nt < 8; ++nt)
      bWih[ks][nt] = load_b_frag(Wih, wave * 128 + nt * 16 + col, DIN, DIN, 512, ks * 32, lane);

  v16h bWhh[4][8];
#pragma unroll
  for (int ks = 0; ks < 4; ++ks)
#pragma unroll
    for (int nt = 0; nt < 8; ++nt)
      bWhh[ks][nt] = load_b_frag(Whh, wave * 128 + nt * 16 + col, HID, HID, 512, ks * 32, lane);

  // FC output fragments: wave w -> N-tile w (d = 16w..16w+15); wave 0 also tile 4 (d=64..79, cols >=72 zero)
  v16h bF0[4], bF1[4];
#pragma unroll
  for (int ks = 0; ks < 4; ++ks) {
    bF0[ks] = load_b_frag(fcW, wave * 16 + col, HID, HID, DIN, ks * 32, lane);
    bF1[ks] = load_b_frag(fcW, 64 + col,        HID, HID, DIN, ks * 32, lane);
  }

  float biasv[8];
#pragma unroll
  for (int nt = 0; nt < 8; ++nt) {
    const int n = wave * 128 + nt * 16 + col;
    biasv[nt] = bih[n] + bhh[n];
  }
  const float fcb0 = (wave * 16 + col < DIN) ? fcb[wave * 16 + col] : 0.0f;
  const float fcb1 = (64 + col < DIN) ? fcb[64 + col] : 0.0f;

  __syncthreads();

  // ================= time loop: 49 encoder + 50 decoder steps =================
  for (int u = 0; u < NSTEPS; ++u) {
    // -- preload ALL A-fragments for this step (deep ds_load window, single drain) --
    v16h aI[3], aH[4];
#pragma unroll
    for (int ks = 0; ks < 3; ++ks) aI[ks] = load_a_frag(&sInp[0][0], KPAD, ks * 32, lane);
#pragma unroll
    for (int ks = 0; ks < 4; ++ks) aH[ks] = load_a_frag(&sH[0][0], HID, ks * 32, lane);

    // prefetch next encoder frame while WMMAs run
    if (u < TIN - 1)
      __builtin_prefetch(&x[((u + 1) * BATCH + row0) * DIN], 0, 1);

    // -- gates = inp @ Wih^T + h @ Whh^T + bias  (each wave: its gate's 128 cols) --
    v8f acc[8];
#pragma unroll
    for (int nt = 0; nt < 8; ++nt)
#pragma unroll
      for (int v = 0; v < 8; ++v) acc[nt][v] = biasv[nt];

#pragma unroll
    for (int ks = 0; ks < 3; ++ks)
#pragma unroll
      for (int nt = 0; nt < 8; ++nt)
        acc[nt] = __builtin_amdgcn_wmma_f32_16x16x32_f16(
            false, aI[ks], false, bWih[ks][nt], (short)0, acc[nt], false, false);
#pragma unroll
    for (int ks = 0; ks < 4; ++ks)
#pragma unroll
      for (int nt = 0; nt < 8; ++nt)
        acc[nt] = __builtin_amdgcn_wmma_f32_16x16x32_f16(
            false, aH[ks], false, bWhh[ks][nt], (short)0, acc[nt], false, false);

    // -- per-gate activation, scatter to LDS (gate order i,f,g,o; g uses tanh) --
    const bool isTanh = (wave == 2);
#pragma unroll
    for (int nt = 0; nt < 8; ++nt)
#pragma unroll
      for (int v = 0; v < 8; ++v) {
        float gv = acc[nt][v];
        gv = isTanh ? tanh_fast(gv) : sigmoid_fast(gv);
        sG[wave][v + mhi][nt * 16 + col] = gv;
      }
    __syncthreads();

    // -- elementwise cell/hidden update: 2048 elems / 128 threads --
#pragma unroll
    for (int e = 0; e < 16; ++e) {
      const int idx = tid * 16 + e;
      const int m = idx >> 7, nc = idx & 127;
      const float cn = sG[1][m][nc] * sC[m][nc] + sG[0][m][nc] * sG[2][m][nc];
      sC[m][nc] = cn;
      sH[m][nc] = (_Float16)(sG[3][m][nc] * tanh_fast(cn));
    }

    // -- encoder: stage next input frame --
    if (u < TIN - 1) {
      for (int i = tid; i < ROWS * DIN; i += 128) {
        const int m = i / DIN, d = i % DIN;
        sInp[m][d] = (_Float16)x[((u + 1) * BATCH + row0 + m) * DIN + d];
      }
    }
    __syncthreads();   // sH complete before FC reads it; sInp reads done before overwrite

    // -- decoder: out = inp + h @ fcW^T + fcb; out becomes next inp --
    if (u >= TIN - 1) {
      const int frame = u + 1;  // 50..99
      v16h aF[4];
#pragma unroll
      for (int ks = 0; ks < 4; ++ks) aF[ks] = load_a_frag(&sH[0][0], HID, ks * 32, lane);

      v8f fa;
#pragma unroll
      for (int v = 0; v < 8; ++v) fa[v] = fcb0;
#pragma unroll
      for (int ks = 0; ks < 4; ++ks)
        fa = __builtin_amdgcn_wmma_f32_16x16x32_f16(
            false, aF[ks], false, bF0[ks], (short)0, fa, false, false);
#pragma unroll
      for (int v = 0; v < 8; ++v) {
        const int m = v + mhi;
        const int d = wave * 16 + col;            // < 64, always valid
        const float o = (float)sInp[m][d] + fa[v];
        sInp[m][d] = (_Float16)o;
        out[(frame * BATCH + row0 + m) * DIN + d] = o;
      }
      if (wave == 0) {                            // tile 4: d = 64..79 (>=72 stays zero)
        v8f fb;
#pragma unroll
        for (int v = 0; v < 8; ++v) fb[v] = fcb1;
#pragma unroll
        for (int ks = 0; ks < 4; ++ks)
          fb = __builtin_amdgcn_wmma_f32_16x16x32_f16(
              false, aF[ks], false, bF1[ks], (short)0, fb, false, false);
#pragma unroll
        for (int v = 0; v < 8; ++v) {
          const int m = v + mhi;
          const int d = 64 + col;
          const float o = (float)sInp[m][d] + fb[v];
          sInp[m][d] = (_Float16)o;               // zero for d>=72 (pad preserved)
          if (d < DIN) out[(frame * BATCH + row0 + m) * DIN + d] = o;
        }
      }
    }
    __syncthreads();
  }
}

// frames 0..49 of the output are a passthrough copy of x
__global__ void copy_x_kernel(const float4* __restrict__ src, float4* __restrict__ dst, int n4) {
  const int i = blockIdx.x * blockDim.x + threadIdx.x;
  if (i < n4) dst[i] = src[i];
}

extern "C" void kernel_launch(void* const* d_in, const int* in_sizes, int n_in,
                              void* d_out, int out_size, void* d_ws, size_t ws_size,
                              hipStream_t stream) {
  (void)in_sizes; (void)n_in; (void)out_size; (void)d_ws; (void)ws_size;
  const float* x   = (const float*)d_in[0];
  const float* Wih = (const float*)d_in[1];
  const float* Whh = (const float*)d_in[2];
  const float* bih = (const float*)d_in[3];
  const float* bhh = (const float*)d_in[4];
  const float* fcW = (const float*)d_in[5];
  const float* fcb = (const float*)d_in[6];
  float* out = (float*)d_out;

  const int n4 = (TIN * BATCH * DIN) / 4;   // 7,372,800 floats -> float4
  copy_x_kernel<<<(n4 + 255) / 256, 256, 0, stream>>>(
      (const float4*)x, (float4*)out, n4);

  lstm_seq2seq_kernel<<<BATCH / ROWS, 128, 0, stream>>>(
      x, Wih, Whh, bih, bhh, fcW, fcb, out);
}
